// Attention_26182120636812
// MI455X (gfx1250) — matrix operand order repaired and checked by reading, not compiled, not run on hardware
//
#include <hip/hip_runtime.h>
#include <hip/hip_bf16.h>

#define B_  2
#define T_  2048
#define D_  2048
#define H_  16
#define KV_ 4
#define HD_ 128
#define REP_ (H_ / KV_)

typedef float        v8f   __attribute__((ext_vector_type(8)));
typedef unsigned int v4u   __attribute__((ext_vector_type(4)));
typedef int          v4i   __attribute__((ext_vector_type(4)));
typedef __bf16       v16bf __attribute__((ext_vector_type(16)));

typedef __attribute__((address_space(1))) v4i* g4p;  // global int4*
typedef __attribute__((address_space(3))) v4i* l4p;  // LDS int4*

union BFrag { unsigned int u[8]; v4u q[2]; v16bf v; };

// ---- CDNA5 async global->LDS copy (ASYNCcnt path), with safe fallback ----
#if defined(__has_builtin)
#if __has_builtin(__builtin_amdgcn_global_load_async_to_lds_b128)
#define ASYNC_CP_OK 1
#endif
#endif
#ifndef ASYNC_CP_OK
#define ASYNC_CP_OK 0
#endif

__device__ __forceinline__ void cp16(const unsigned short* g, unsigned short* l) {
#if ASYNC_CP_OK
  __builtin_amdgcn_global_load_async_to_lds_b128((g4p)(unsigned short*)g,
                                                 (l4p)l, 0, 0);
#else
  *(v4u*)l = *(const v4u*)g;
#endif
}

__device__ __forceinline__ void async_wait0() {
#if ASYNC_CP_OK
#if __has_builtin(__builtin_amdgcn_s_wait_asynccnt)
  __builtin_amdgcn_s_wait_asynccnt(0);
#else
  asm volatile("s_wait_asynccnt 0x0" ::: "memory");
#endif
#endif
}

__device__ __forceinline__ unsigned short f2bf(float f) {
  unsigned int u = __float_as_uint(f);
  u += 0x7FFFu + ((u >> 16) & 1u);   // round-to-nearest-even
  return (unsigned short)(u >> 16);
}

__global__ void cvt_f32_bf16(const float* __restrict__ in,
                             unsigned short* __restrict__ out, int n) {
  int i = blockIdx.x * blockDim.x + threadIdx.x;
  if (i < n) out[i] = f2bf(in[i]);
}

// out[c*R + r] = bf16(in[r*C + c]); 16x16 LDS-tiled, block = 256 threads.
__global__ void transpose_f32_bf16(const float* __restrict__ in,
                                   unsigned short* __restrict__ out,
                                   int R, int C) {
  __shared__ float tile[16][17];
  const int r0 = blockIdx.y * 16, c0 = blockIdx.x * 16;
  const int tx = threadIdx.x & 15, ty = threadIdx.x >> 4;
  tile[ty][tx] = in[(size_t)(r0 + ty) * C + c0 + tx];
  __syncthreads();
  out[(size_t)(c0 + ty) * R + r0 + tx] = f2bf(tile[tx][ty]);
}

// C[M,N] fp32 = A[M,K] @ B[K,N]; A row-major bf16, Bt = B transposed (N x K) bf16.
// Block: 128 threads (4 waves), 64x64 tile, K step 32, HBM->LDS async staging.
#define LDSP 40   // LDS row pitch in ushorts (80B): conflict-free, 16B aligned
__global__ __launch_bounds__(128) void gemm_bf16_lds(
    const unsigned short* __restrict__ A,
    const unsigned short* __restrict__ Bt,
    float* __restrict__ C, int M, int N, int K) {
  __shared__ unsigned short As[64 * LDSP];
  __shared__ unsigned short Bs[64 * LDSP];
  const int tid  = threadIdx.x;
  const int w    = tid >> 5;          // wave id: M sub-tile
  const int lane = tid & 31;
  const int m  = lane & 15;
  const int hf = lane >> 4;
  const int mt = blockIdx.y * 64, nt = blockIdx.x * 64;

  v8f acc[4] = {};
  for (int k0 = 0; k0 < K; k0 += 32) {
    // stage A(64x32) and Bt(64x32) tiles: 256 x 16B segments each, 128 threads
#pragma unroll
    for (int i = 0; i < 2; ++i) {
      const int s   = tid + i * 128;
      const int row = s >> 2, seg = s & 3;
      cp16(A  + (size_t)(mt + row) * K + k0 + seg * 8, As + row * LDSP + seg * 8);
      cp16(Bt + (size_t)(nt + row) * K + k0 + seg * 8, Bs + row * LDSP + seg * 8);
    }
    async_wait0();
    __syncthreads();

    BFrag a;
    const unsigned short* ap = As + (w * 16 + m) * LDSP;
    a.q[0] = *(const v4u*)(ap + hf * 8);        // K = hf*8   .. +7
    a.q[1] = *(const v4u*)(ap + 16 + hf * 8);   // K = 16+hf*8.. +7
#pragma unroll
    for (int nt4 = 0; nt4 < 4; ++nt4) {
      BFrag b;
      const unsigned short* bp = Bs + (nt4 * 16 + m) * LDSP;  // lane = N col
      b.q[0] = *(const v4u*)(bp + hf * 8);      // K = hf*8   .. +7
      b.q[1] = *(const v4u*)(bp + 16 + hf * 8); // K = 16+hf*8.. +7
      acc[nt4] = __builtin_amdgcn_wmma_f32_16x16x32_bf16(false, a.v, false, b.v,
                                                         (short)0, acc[nt4], false, false);
    }
    __syncthreads();
  }
#pragma unroll
  for (int nt4 = 0; nt4 < 4; ++nt4)
#pragma unroll
    for (int j = 0; j < 8; ++j)
      C[(size_t)(mt + w * 16 + j + 8 * hf) * N + nt + nt4 * 16 + m] = acc[nt4][j];
}

// Fused RMSNorm + RoPE; one wave per (row, head); fp32 in -> bf16 out.
__global__ void rmsnorm_rope_bf16(const float* __restrict__ in,
                                  const float* __restrict__ w,
                                  const float* __restrict__ fcos,
                                  const float* __restrict__ fsin,
                                  unsigned short* __restrict__ out, int nheads) {
  const int lane = threadIdx.x;
  const int row  = blockIdx.x;            // b*T + t
  const int h    = blockIdx.y;
  const int t    = row & (T_ - 1);
  const size_t base = (size_t)row * ((size_t)nheads * HD_) + (size_t)h * HD_ + lane * 4;
  float u0 = in[base], u1 = in[base + 1], u2 = in[base + 2], u3 = in[base + 3];
  float ss = u0 * u0 + u1 * u1 + u2 * u2 + u3 * u3;
#pragma unroll
  for (int d = 1; d < 32; d <<= 1) ss += __shfl_xor(ss, d, 32);
  const float r = rsqrtf(ss * (1.0f / HD_) + 1e-6f);
  const int hd = lane * 4;
  u0 *= r * w[hd];     u1 *= r * w[hd + 1];
  u2 *= r * w[hd + 2]; u3 *= r * w[hd + 3];
  const int i0 = lane * 2, i1 = i0 + 1;
  const float c0 = fcos[t * (HD_ / 2) + i0], s0 = fsin[t * (HD_ / 2) + i0];
  const float c1 = fcos[t * (HD_ / 2) + i1], s1 = fsin[t * (HD_ / 2) + i1];
  out[base]     = f2bf(u0 * c0 - u1 * s0);
  out[base + 1] = f2bf(u0 * s0 + u1 * c0);
  out[base + 2] = f2bf(u2 * c1 - u3 * s1);
  out[base + 3] = f2bf(u2 * s1 + u3 * c1);
}

// Flash-style causal GQA attention. One wave per (b, head, 16-query tile).
// Q/K row-major bf16; Vt transposed bf16 [(b*KV+g)*HD + hd][T].
__global__ void flash_attn(const unsigned short* __restrict__ Q,
                           const unsigned short* __restrict__ Kb,
                           const unsigned short* __restrict__ Vt,
                           float* __restrict__ O) {
  __shared__ unsigned short Pl[16][32];
  const int lane = threadIdx.x;
  const int m  = lane & 15;
  const int hf = lane >> 4;
  const int m0 = blockIdx.x * 16;
  const int h  = blockIdx.y;
  const int b  = blockIdx.z;
  const int g  = h / REP_;
  const float scale = 0.08838834764831845f;  // 1/sqrt(128)

  BFrag qa[4];   // Q A-frags: 16x128 as four 16x32 chunks
  {
    const size_t qrow = ((size_t)(b * T_ + m0 + m)) * (H_ * HD_) + (size_t)h * HD_;
#pragma unroll
    for (int c = 0; c < 4; ++c) {
      qa[c].q[0] = *(const v4u*)(Q + qrow + c * 32 + hf * 8);
      qa[c].q[1] = *(const v4u*)(Q + qrow + c * 32 + 16 + hf * 8);
    }
  }

  v8f acc[8] = {};
  float mrun[8], lrun[8];
#pragma unroll
  for (int j = 0; j < 8; ++j) { mrun[j] = -3.0e38f; lrun[j] = 0.0f; }

  const int nkb = (m0 + 47) >> 5;
  for (int kb = 0; kb < nkb; ++kb) {
    const int kbase = kb * 32;
    v8f st[2] = {};
#pragma unroll
    for (int tile = 0; tile < 2; ++tile) {
      const int n0 = kbase + tile * 16;
      const size_t krow = ((size_t)(b * T_ + n0 + m)) * (KV_ * HD_) + (size_t)g * HD_;
      v8f s = {};
#pragma unroll
      for (int c = 0; c < 4; ++c) {
        BFrag bk;
        bk.q[0] = *(const v4u*)(Kb + krow + c * 32 + hf * 8);
        bk.q[1] = *(const v4u*)(Kb + krow + c * 32 + 16 + hf * 8);
        s = __builtin_amdgcn_wmma_f32_16x16x32_bf16(false, qa[c].v, false, bk.v,
                                                    (short)0, s, false, false);
      }
      st[tile] = s;
    }

    float alpha[8];
#pragma unroll
    for (int j = 0; j < 8; ++j) {
      const int qrowg = m0 + j + 8 * hf;
      float v0 = st[0][j] * scale;
      float v1 = st[1][j] * scale;
      if (kbase + m > qrowg)      v0 = -1.0e30f;
      if (kbase + 16 + m > qrowg) v1 = -1.0e30f;
      float rmax = fmaxf(v0, v1);
#pragma unroll
      for (int d = 1; d <= 8; d <<= 1) rmax = fmaxf(rmax, __shfl_xor(rmax, d, 32));
      const float mnew = fmaxf(mrun[j], rmax);
      alpha[j] = __expf(mrun[j] - mnew);
      mrun[j]  = mnew;
      const float p0 = __expf(v0 - mnew);
      const float p1 = __expf(v1 - mnew);
      float rs = p0 + p1;
#pragma unroll
      for (int d = 1; d <= 8; d <<= 1) rs += __shfl_xor(rs, d, 32);
      lrun[j] = lrun[j] * alpha[j] + rs;
      Pl[j + 8 * hf][m]      = f2bf(p0);
      Pl[j + 8 * hf][m + 16] = f2bf(p1);
    }
    __syncthreads();

    BFrag pa;
    pa.q[0] = *(const v4u*)&Pl[m][hf * 8];
    pa.q[1] = *(const v4u*)&Pl[m][16 + hf * 8];

#pragma unroll
    for (int nt = 0; nt < 8; ++nt)
#pragma unroll
      for (int j = 0; j < 8; ++j) acc[nt][j] *= alpha[j];

#pragma unroll
    for (int nt = 0; nt < 8; ++nt) {
      BFrag bv;  // V B-frag from transposed V: lane = hd col, contiguous keys
      const size_t vrow = ((size_t)(b * (KV_ * HD_) + g * HD_ + nt * 16 + m)) * T_;
      bv.q[0] = *(const v4u*)(Vt + vrow + kbase + hf * 8);
      bv.q[1] = *(const v4u*)(Vt + vrow + kbase + 16 + hf * 8);
      acc[nt] = __builtin_amdgcn_wmma_f32_16x16x32_bf16(false, pa.v, false, bv.v,
                                                        (short)0, acc[nt], false, false);
    }
    __syncthreads();
  }

#pragma unroll
  for (int nt = 0; nt < 8; ++nt)
#pragma unroll
    for (int j = 0; j < 8; ++j) {
      const size_t o = ((size_t)(b * T_ + m0 + j + 8 * hf)) * (H_ * HD_) +
                       (size_t)h * HD_ + nt * 16 + m;
      O[o] = acc[nt][j] / lrun[j];
    }
}

extern "C" void kernel_launch(void* const* d_in, const int* in_sizes, int n_in,
                              void* d_out, int out_size, void* d_ws, size_t ws_size,
                              hipStream_t stream) {
  const float* x   = (const float*)d_in[0];
  const float* wq  = (const float*)d_in[1];
  const float* wk  = (const float*)d_in[2];
  const float* wv  = (const float*)d_in[3];
  const float* wo  = (const float*)d_in[4];
  const float* qnw = (const float*)d_in[5];
  const float* knw = (const float*)d_in[6];
  const float* fc  = (const float*)d_in[7];
  const float* fs  = (const float*)d_in[8];
  float* out = (float*)d_out;

  char* ws = (char*)d_ws;
  size_t off = 0;
  auto alloc = [&](size_t bytes) {
    char* p = ws + off;
    off += (bytes + 255) & ~(size_t)255;
    return p;
  };

  const size_t BT = (size_t)B_ * T_;  // 4096
  unsigned short* xb   = (unsigned short*)alloc(BT * D_ * 2);
  unsigned short* wqT  = (unsigned short*)alloc((size_t)D_ * H_ * HD_ * 2);   // (H*HD) x D
  unsigned short* wkT  = (unsigned short*)alloc((size_t)D_ * KV_ * HD_ * 2);  // (KV*HD) x D
  unsigned short* wvT  = (unsigned short*)alloc((size_t)D_ * KV_ * HD_ * 2);
  unsigned short* woT  = (unsigned short*)alloc((size_t)H_ * HD_ * D_ * 2);   // D x (H*HD)
  float* qf = (float*)alloc(BT * H_ * HD_ * 4);
  float* kf = (float*)alloc(BT * KV_ * HD_ * 4);
  float* vf = (float*)alloc(BT * KV_ * HD_ * 4);
  unsigned short* qb = (unsigned short*)alloc(BT * H_ * HD_ * 2);
  unsigned short* kb = (unsigned short*)alloc(BT * KV_ * HD_ * 2);
  unsigned short* vt = (unsigned short*)alloc(BT * KV_ * HD_ * 2);  // transposed V
  float* aof          = qf;   // alias: qf dead after rmsnorm_rope(q)
  unsigned short* aob = xb;   // alias: xb dead after the V projection GEMM

  auto cvt = [&](const float* src, unsigned short* dst, size_t n) {
    cvt_f32_bf16<<<dim3((unsigned)((n + 255) / 256)), dim3(256), 0, stream>>>(src, dst, (int)n);
  };
  auto tp = [&](const float* src, unsigned short* dst, int R, int C) {
    transpose_f32_bf16<<<dim3(C / 16, R / 16), dim3(256), 0, stream>>>(src, dst, R, C);
  };

  // 1) bf16 conversion; weights transposed to N x K for b128 fragment loads
  cvt(x, xb, BT * D_);
  tp(wq, wqT, D_, H_ * HD_);
  tp(wk, wkT, D_, KV_ * HD_);
  tp(wv, wvT, D_, KV_ * HD_);
  tp(wo, woT, H_ * HD_, D_);

  // 2) QKV projections (LDS-tiled WMMA, async HBM->LDS staging)
  gemm_bf16_lds<<<dim3((H_ * HD_) / 64, (unsigned)(BT / 64)), dim3(128), 0, stream>>>(
      xb, wqT, qf, (int)BT, H_ * HD_, D_);
  gemm_bf16_lds<<<dim3((KV_ * HD_) / 64, (unsigned)(BT / 64)), dim3(128), 0, stream>>>(
      xb, wkT, kf, (int)BT, KV_ * HD_, D_);
  gemm_bf16_lds<<<dim3((KV_ * HD_) / 64, (unsigned)(BT / 64)), dim3(128), 0, stream>>>(
      xb, wvT, vf, (int)BT, KV_ * HD_, D_);

  // 3) V -> transposed bf16 [(b*KV+g)*HD + hd][T]
  tp(vf,                          vt,                          T_, KV_ * HD_);  // b = 0
  tp(vf + (size_t)T_ * KV_ * HD_, vt + (size_t)T_ * KV_ * HD_, T_, KV_ * HD_);  // b = 1

  // 4) RMSNorm + RoPE on Q and K -> bf16
  rmsnorm_rope_bf16<<<dim3((unsigned)BT, H_), dim3(32), 0, stream>>>(qf, qnw, fc, fs, qb, H_);
  rmsnorm_rope_bf16<<<dim3((unsigned)BT, KV_), dim3(32), 0, stream>>>(kf, knw, fc, fs, kb, KV_);

  // 5) Causal GQA flash attention (WMMA)
  flash_attn<<<dim3(T_ / 16, H_, B_), dim3(32), 0, stream>>>(qb, kb, vt, aof);

  // 6) Output projection (WMMA)
  cvt(aof, aob, BT * H_ * HD_);
  gemm_bf16_lds<<<dim3(D_ / 64, (unsigned)(BT / 64)), dim3(128), 0, stream>>>(
      aob, woT, out, (int)BT, D_, H_ * HD_);

  (void)in_sizes; (void)n_in; (void)out_size; (void)ws_size;
}
